// WorkingMemoryController_64338610094845
// MI455X (gfx1250) — compile-verified, matrix operand order checked by reading, not run on hardware
//
#include <hip/hip_runtime.h>
#include <hip/hip_bf16.h>
#include <math.h>

// ---------------------------------------------------------------------------
// DNC-style WorkingMemoryController for MI455X (gfx1250, wave32, WMMA)
// Controller GEMM now uses async global->LDS weight staging (ASYNCcnt path)
// with a double-buffered LDS slab, overlapping L2 fetch with WMMA compute.
// ---------------------------------------------------------------------------

typedef __bf16 bf16;
typedef __attribute__((ext_vector_type(16))) __bf16 v16bf;
typedef __attribute__((ext_vector_type(8)))  __bf16 v8bf;
typedef __attribute__((ext_vector_type(8)))  float  v8f;

#define DEVINL __device__ __forceinline__

// ---- model dims -----------------------------------------------------------
constexpr int Wd    = 128;
constexpr int Rr    = 4;
constexpr int Nn    = 128;
constexpr int Bb    = 64;
constexpr int Tt    = 1024;
constexpr int IND   = (Rr + 1) * Wd;                 // 640
constexpr int IFACE = Wd * Rr + 3 * Wd + 5 * Rr + 3; // 919
constexpr int OUTD  = Wd + IFACE;                    // 1047
// padded dims (WMMA-friendly)
constexpr int NPC = 1056;          // 66 * 16  (padded OUT)
constexpr int KPC = IND + NPC;     // 1696 = 53 * 32 (padded controller K)
constexpr int KPE = 256;           // encoder K = 128(x) + 128(h)
constexpr int KITER = KPC / 32;    // 53

// ---- math helpers ---------------------------------------------------------
DEVINL float sigmoidf(float x) { return 1.0f / (1.0f + __expf(-x)); }
DEVINL float tanh_fast(float x) {
  if (x >  10.0f) return  1.0f;
  if (x < -10.0f) return -1.0f;
  float e = __expf(2.0f * x);
  return (e - 1.0f) / (e + 1.0f);
}
DEVINL float oneplus_f(float z) {
  return 1.0f + ((z > 20.0f) ? z : __logf(1.0f + __expf(z)));
}

// ---- CDNA5 async global->LDS copy (ASYNCcnt) ------------------------------
// GVS mode: mem_addr = SADDR(64) + VADDR[31:0]; per-lane 16B into per-lane
// LDS byte address held in the first VGPR operand.
DEVINL void async_g2l_b128(unsigned lds_byte, unsigned g_off, unsigned long long sbase) {
  asm volatile("global_load_async_to_lds_b128 %0, %1, %2"
               :: "v"(lds_byte), "v"(g_off), "s"(sbase)
               : "memory");
}
DEVINL void wait_async0() {
#if __has_builtin(__builtin_amdgcn_s_wait_asynccnt)
  __builtin_amdgcn_s_wait_asynccnt(0);
#else
  asm volatile("s_wait_asynccnt 0x0" ::: "memory");
#endif
}

// ---- WMMA fragment helpers ------------------------------------------------
DEVINL v16bf load_frag_rm(const bf16* __restrict__ base, int stride,
                          int row0, int k0, int lane) {
  int half = lane >> 4;
  int r    = lane & 15;
  const bf16* p = base + (size_t)(row0 + r) * stride + k0 + 8 * half;
  union { v16bf v; v8bf h[2]; } u;
  u.h[0] = *(const v8bf*)(p);
  u.h[1] = *(const v8bf*)(p + 16);
  return u.v;
}

DEVINL v16bf load_frag_lds(const bf16* Bs, int rbase, int lane) {
  int half = lane >> 4;
  int r    = lane & 15;
  const bf16* p = Bs + (rbase + r) * 32 + 8 * half;
  union { v16bf v; v8bf h[2]; } u;
  u.h[0] = *(const v8bf*)(p);
  u.h[1] = *(const v8bf*)(p + 16);
  return u.v;
}

DEVINL v8f wmma_bf16(v16bf a, v16bf b, v8f c) {
  return __builtin_amdgcn_wmma_f32_16x16x32_bf16(
      /*neg_a=*/false, a, /*neg_b=*/false, b,
      /*c_mod=*/(short)0, c, /*reuse_a=*/false, /*reuse_b=*/false);
}

// ---- block reductions (128 threads) --------------------------------------
DEVINL float block_sum(float v, float* red, int tid) {
  red[tid] = v; __syncthreads();
  #pragma unroll
  for (int s = 64; s > 0; s >>= 1) {
    if (tid < s) red[tid] += red[tid + s];
    __syncthreads();
  }
  float r = red[0]; __syncthreads();
  return r;
}
DEVINL float block_max(float v, float* red, int tid) {
  red[tid] = v; __syncthreads();
  #pragma unroll
  for (int s = 64; s > 0; s >>= 1) {
    if (tid < s) { float o = red[tid + s]; if (o > red[tid]) red[tid] = o; }
    __syncthreads();
  }
  float r = red[0]; __syncthreads();
  return r;
}

// ===========================================================================
// Prologue kernels
// ===========================================================================
__global__ void fill_f32(float* p, size_t n, float v) {
  for (size_t i = (size_t)blockIdx.x * blockDim.x + threadIdx.x; i < n;
       i += (size_t)gridDim.x * blockDim.x)
    p[i] = v;
}
__global__ void copy_f32(const float* __restrict__ s, float* __restrict__ d, size_t n) {
  for (size_t i = (size_t)blockIdx.x * blockDim.x + threadIdx.x; i < n;
       i += (size_t)gridDim.x * blockDim.x)
    d[i] = s[i];
}

// encoder weights: Wenc[512][256] = [Wih | Whh] in bf16
__global__ void prep_enc_w(const float* __restrict__ Wih, const float* __restrict__ Whh,
                           const float* __restrict__ bsrc,
                           bf16* __restrict__ Wenc, float* __restrict__ benc) {
  int i = blockIdx.x * blockDim.x + threadIdx.x;
  if (i < 512 * 256) {
    int row = i >> 8, col = i & 255;
    float v = (col < Wd) ? Wih[row * Wd + col] : Whh[row * Wd + (col - Wd)];
    Wenc[i] = (bf16)v;
  }
  if (i < 512) benc[i] = bsrc[i];
}

// controller weights: Wc[4][NPC][KPC] = [Wih | Whh | 0pad] per gate
__global__ void prep_ctl_w(const float* __restrict__ Wih, const float* __restrict__ Whh,
                           bf16* __restrict__ Wc) {
  const size_t total = (size_t)4 * NPC * KPC;
  for (size_t i = (size_t)blockIdx.x * blockDim.x + threadIdx.x; i < total;
       i += (size_t)gridDim.x * blockDim.x) {
    int    k   = (int)(i % KPC);
    size_t rem = i / KPC;
    int    n   = (int)(rem % NPC);
    int    q   = (int)(rem / NPC);
    float v = 0.0f;
    if (n < OUTD) {
      int row = q * OUTD + n;
      if (k < IND)             v = Wih[(size_t)row * IND + k];
      else if (k < IND + OUTD) v = Whh[(size_t)row * OUTD + (k - IND)];
    }
    Wc[i] = (bf16)v;
  }
}

__global__ void prep_ctl_b(const float* __restrict__ bsrc, float* __restrict__ bc) {
  int i = blockIdx.x * blockDim.x + threadIdx.x;
  if (i < 4 * NPC) {
    int n = i % NPC, q = i / NPC;
    bc[i] = (n < OUTD) ? bsrc[q * OUTD + n] : 0.0f;
  }
}

// x (B,T,W) f32 -> xbf (T,B,W) bf16 (time-major for encoder steps)
__global__ void prep_xbf(const float* __restrict__ x, bf16* __restrict__ xbf) {
  const size_t total = (size_t)Bb * Tt * Wd;
  for (size_t i = (size_t)blockIdx.x * blockDim.x + threadIdx.x; i < total;
       i += (size_t)gridDim.x * blockDim.x) {
    int w = (int)(i % Wd);
    int t = (int)((i / Wd) % Tt);
    int b = (int)(i / ((size_t)Wd * Tt));
    xbf[((size_t)t * Bb + b) * Wd + w] = (bf16)x[i];
  }
}

// act[0] input region: interleave enc[0] with rv=0
__global__ void ctl_init_act(const bf16* __restrict__ encs0, bf16* __restrict__ act0) {
  int b = blockIdx.x, w = threadIdx.x;
  bf16* ap = act0 + (size_t)b * KPC + w * (Rr + 1);
  ap[0] = encs0[b * Wd + w];
  #pragma unroll
  for (int r = 0; r < Rr; r++) ap[1 + r] = (bf16)0.0f;
}

// ===========================================================================
// Encoder LSTM step: g = [x_t | h] @ Wenc^T + b ; 8 WGs x 4 waves, WMMA bf16
// ===========================================================================
__global__ void enc_step_kernel(const bf16* __restrict__ xt,
                                const bf16* __restrict__ hin,
                                const bf16* __restrict__ Wenc,
                                const float* __restrict__ benc,
                                float* __restrict__ cenc,
                                bf16* __restrict__ hout,
                                bf16* __restrict__ encs_t) {
  int lane  = threadIdx.x & 31;
  int mtile = threadIdx.x >> 5;   // 0..3 -> batch tiles
  int n0    = blockIdx.x * 16;    // hidden column tile
  int m0    = mtile * 16;

  v8f acc[4] = {};
  #pragma unroll
  for (int kk = 0; kk < 8; kk++) {
    const bf16* abase = (kk < 4) ? xt : hin;
    int k0a = (kk < 4) ? kk * 32 : (kk - 4) * 32;
    v16bf a = load_frag_rm(abase, Wd, m0, k0a, lane);
    #pragma unroll
    for (int q = 0; q < 4; q++) {
      v16bf bfr = load_frag_rm(Wenc, KPE, q * Wd + n0, kk * 32, lane);
      acc[q] = wmma_bf16(a, bfr, acc[q]);
    }
  }

  int half = lane >> 4, nr = lane & 15;
  int n = n0 + nr;
  #pragma unroll
  for (int r = 0; r < 8; r++) {
    int m = m0 + half * 8 + r;
    float gi = acc[0][r] + benc[0 * Wd + n];
    float gf = acc[1][r] + benc[1 * Wd + n];
    float gg = acc[2][r] + benc[2 * Wd + n];
    float go = acc[3][r] + benc[3 * Wd + n];
    float c  = cenc[m * Wd + n];
    c = sigmoidf(gf) * c + sigmoidf(gi) * tanh_fast(gg);
    float h = sigmoidf(go) * tanh_fast(c);
    cenc[m * Wd + n]   = c;
    hout[m * Wd + n]   = (bf16)h;
    encs_t[m * Wd + n] = (bf16)h;
  }
}

// ===========================================================================
// Controller LSTM step (fused GEMM + gates), async-LDS double-buffered weights
// grid: 66 WGs (column tiles), 4 waves = 4 batch tiles.
// Per k-iter: wait_async -> barrier -> kick next slab (other buffer) -> WMMA.
// ===========================================================================
__global__ void ctl_step_kernel(const bf16* __restrict__ act_in,  // [B][KPC]
                                const bf16* __restrict__ Wc,      // [4][NPC][KPC]
                                const float* __restrict__ bc,     // [4][NPC]
                                float* __restrict__ cst,          // [B][NPC]
                                float* __restrict__ hfull,        // [B][NPC]
                                bf16* __restrict__ act_out,       // [B][KPC]
                                float* __restrict__ out,          // [B][T][W]
                                int t) {
  __shared__ bf16 Bs[2][64 * 32];  // double-buffered 4KB weight slab

  int tid   = threadIdx.x;
  int lane  = tid & 31;
  int mtile = tid >> 5;          // 0..3
  int n0    = blockIdx.x * 16;
  int m0    = mtile * 16;

  v8f acc[4] = {};

  // this thread's 32B chunk of the 64x32 slab (128 thr x 32B = 4KB)
  int lrow = tid >> 1;                 // 0..63 : gate-major weight row
  int lofs = (tid & 1) * 16;           // element offset 0/16 within slab row
  int lq   = lrow >> 4;
  int ln   = n0 + (lrow & 15);

  const unsigned long long wbase = (unsigned long long)(uintptr_t)Wc;
  const unsigned gbyte = (unsigned)((((size_t)(lq * NPC + ln)) * KPC + lofs) * 2);
  const unsigned lchunk = (unsigned)((lrow * 32 + lofs) * 2);
  const unsigned lds0 = (unsigned)(uintptr_t)(&Bs[0][0]);

  // preload slab 0 into buffer 0 (two async 16B pieces per thread)
  async_g2l_b128(lds0 + lchunk,      gbyte,      wbase);
  async_g2l_b128(lds0 + lchunk + 16, gbyte + 16, wbase);

  for (int kk = 0; kk < KITER; kk++) {
    wait_async0();
    __syncthreads();

    int cur = kk & 1;
    if (kk + 1 < KITER) {
      unsigned g = gbyte + (unsigned)((kk + 1) * 64);  // 32 elems * 2B
      unsigned l = lds0 + (unsigned)((cur ^ 1) * (64 * 32 * 2)) + lchunk;
      async_g2l_b128(l,      g,      wbase);
      async_g2l_b128(l + 16, g + 16, wbase);
    }

    v16bf a = load_frag_rm(act_in, KPC, m0, kk * 32, lane);
    #pragma unroll
    for (int q = 0; q < 4; q++) {
      v16bf bfr = load_frag_lds(&Bs[cur][0], q * 16, lane);
      acc[q] = wmma_bf16(a, bfr, acc[q]);
    }
  }

  int half = lane >> 4, nr = lane & 15;
  int n = n0 + nr;
  if (n < OUTD) {
    #pragma unroll
    for (int r = 0; r < 8; r++) {
      int m = m0 + half * 8 + r;    // batch index
      float gi = acc[0][r] + bc[0 * NPC + n];
      float gf = acc[1][r] + bc[1 * NPC + n];
      float gg = acc[2][r] + bc[2 * NPC + n];
      float go = acc[3][r] + bc[3 * NPC + n];
      float c  = cst[m * NPC + n];
      c = sigmoidf(gf) * c + sigmoidf(gi) * tanh_fast(gg);
      float h = sigmoidf(go) * tanh_fast(c);
      cst[m * NPC + n]   = c;
      hfull[m * NPC + n] = h;
      act_out[(size_t)m * KPC + IND + n] = (bf16)h;
      if (n < Wd)
        out[((size_t)m * Tt + t) * Wd + n] = h;
    }
  }
}

// ===========================================================================
// Memory step: cosine addressing, softmax over N, M update, read vectors,
// build next activation's input region. One WG (128 thr) per batch element.
// ===========================================================================
__global__ void mem_step_kernel(const float* __restrict__ hfull, // [B][NPC]
                                float* __restrict__ M,           // [B][N][W]
                                const bf16* __restrict__ encs_next, // [B][W]
                                bf16* __restrict__ act_out,      // [B][KPC]
                                int build_next) {
  int b = blockIdx.x, tid = threadIdx.x;  // tid: 0..127 (= n or w)
  __shared__ float kw[128], ev[128], vv[128];
  __shared__ float kr[4][128];
  __shared__ float dots[4][128], nrm[128], wrs[4][128];
  __shared__ float red[128];
  __shared__ float sc[8];  // [0..3]=br, [4]=bw, [5]=gw

  const float* xi = hfull + (size_t)b * NPC + Wd;  // interface vector
  // layout: kr[0:512], br[512:516], kw[516:644], bw[644], e[645:773],
  //         v[773:901], (skip R+1), gw[906]
  kw[tid] = xi[516 + tid];
  ev[tid] = sigmoidf(xi[645 + tid]);
  vv[tid] = xi[773 + tid];
  #pragma unroll
  for (int r = 0; r < 4; r++) kr[r][tid] = xi[r * Wd + tid];
  if (tid < 4)       sc[tid] = oneplus_f(xi[512 + tid]);
  else if (tid == 4) sc[4]   = oneplus_f(xi[644]);
  else if (tid == 5) sc[5]   = sigmoidf(xi[906]);
  __syncthreads();

  // key norms (redundantly per-thread; tiny)
  float kwn = 0.0f, krn[4] = {0, 0, 0, 0};
  for (int w = 0; w < Wd; w++) {
    kwn += kw[w] * kw[w];
    #pragma unroll
    for (int r = 0; r < 4; r++) krn[r] += kr[r][w] * kr[r][w];
  }
  kwn = sqrtf(kwn) + 1e-8f;
  #pragma unroll
  for (int r = 0; r < 4; r++) krn[r] = sqrtf(krn[r]) + 1e-8f;

  // --- write addressing: gw * softmax_n(bw * cos(M[n], kw)) ---
  float* Mb = M + (size_t)b * Nn * Wd;
  float* Mr = Mb + tid * Wd;  // thread owns memory row n = tid
  float dot = 0.0f, mn2 = 0.0f;
  for (int w = 0; w < Wd; w++) { float m = Mr[w]; dot += m * kw[w]; mn2 += m * m; }
  float s  = sc[4] * dot / ((sqrtf(mn2) + 1e-8f) * kwn);
  float mx = block_max(s, red, tid);
  float e0 = __expf(s - mx);
  float sm = block_sum(e0, red, tid);
  float ww = sc[5] * e0 / sm;

  // --- memory update for row n=tid, fused with new norms / kr dots ---
  float nn2 = 0.0f, dr[4] = {0, 0, 0, 0};
  for (int w = 0; w < Wd; w++) {
    float m = Mr[w];
    m = m * (1.0f - ww * ev[w]) + ww * vv[w];
    Mr[w] = m;
    nn2 += m * m;
    #pragma unroll
    for (int r = 0; r < 4; r++) dr[r] += m * kr[r][w];
  }
  nrm[tid] = sqrtf(nn2) + 1e-8f;
  #pragma unroll
  for (int r = 0; r < 4; r++) dots[r][tid] = dr[r];
  __syncthreads();

  // --- read addressing: softmax_n(br_r * cos(M'[n], kr_r)) ---
  for (int r = 0; r < 4; r++) {
    float sr  = sc[r] * dots[r][tid] / (nrm[tid] * krn[r]);
    float mxr = block_max(sr, red, tid);
    float er  = __expf(sr - mxr);
    float smr = block_sum(er, red, tid);
    wrs[r][tid] = er / smr;
  }
  __syncthreads();

  // --- read vectors rv[w=tid][r] = sum_n wr[n][r] * M'[n][w] ---
  float rv[4] = {0, 0, 0, 0};
  for (int n2 = 0; n2 < Nn; n2++) {
    float m = Mb[n2 * Wd + tid];
    #pragma unroll
    for (int r = 0; r < 4; r++) rv[r] += wrs[r][n2] * m;
  }

  if (build_next) {
    bf16* ap = act_out + (size_t)b * KPC + tid * (Rr + 1);
    ap[0] = encs_next[(size_t)b * Wd + tid];
    #pragma unroll
    for (int r = 0; r < 4; r++) ap[1 + r] = (bf16)rv[r];
  }
}

// ===========================================================================
// Host launch
// ===========================================================================
extern "C" void kernel_launch(void* const* d_in, const int* in_sizes, int n_in,
                              void* d_out, int out_size, void* d_ws, size_t ws_size,
                              hipStream_t stream) {
  (void)in_sizes; (void)n_in; (void)out_size; (void)ws_size;
  const float* x      = (const float*)d_in[0];
  // d_in[1] = source_lengths (all == T) -- unused
  const float* encWih = (const float*)d_in[2];
  const float* encWhh = (const float*)d_in[3];
  const float* encB   = (const float*)d_in[4];
  const float* ctlWih = (const float*)d_in[5];
  const float* ctlWhh = (const float*)d_in[6];
  const float* ctlB   = (const float*)d_in[7];
  const float* M0     = (const float*)d_in[8];
  float* out = (float*)d_out;

  // ---- workspace carve (256B aligned) ----
  char* ws = (char*)d_ws;
  size_t off = 0;
  auto carve = [&](size_t bytes) -> char* {
    char* p = ws + off;
    off = (off + bytes + 255) & ~(size_t)255;
    return p;
  };
  bf16*  Wenc = (bf16*) carve((size_t)512 * KPE * 2);
  float* benc = (float*)carve(512 * 4);
  bf16*  Wc   = (bf16*) carve((size_t)4 * NPC * KPC * 2);     // 14.3 MB
  float* bc   = (float*)carve((size_t)4 * NPC * 4);
  bf16*  xbf  = (bf16*) carve((size_t)Tt * Bb * Wd * 2);      // 16 MB (T,B,W)
  bf16*  encs = (bf16*) carve((size_t)Tt * Bb * Wd * 2);      // 16 MB (T,B,W)
  bf16*  henc = (bf16*) carve((size_t)2 * Bb * Wd * 2);
  float* cenc = (float*)carve((size_t)Bb * Wd * 4);
  bf16*  act  = (bf16*) carve((size_t)2 * Bb * KPC * 2);
  float* cst  = (float*)carve((size_t)Bb * NPC * 4);
  float* hful = (float*)carve((size_t)Bb * NPC * 4);
  float* Mw   = (float*)carve((size_t)Bb * Nn * Wd * 4);      // 4 MB

  // ---- prologue: weight repack (fp32 -> padded bf16), state init ----
  prep_enc_w<<<(512 * 256 + 255) / 256, 256, 0, stream>>>(encWih, encWhh, encB, Wenc, benc);
  prep_ctl_w<<<4096, 256, 0, stream>>>(ctlWih, ctlWhh, Wc);
  prep_ctl_b<<<(4 * NPC + 255) / 256, 256, 0, stream>>>(ctlB, bc);
  prep_xbf<<<4096, 256, 0, stream>>>(x, xbf);

  fill_f32<<<64, 256, 0, stream>>>(cenc, (size_t)Bb * Wd, 0.0f);
  fill_f32<<<64, 256, 0, stream>>>((float*)henc, (size_t)2 * Bb * Wd / 2, 0.0f);
  fill_f32<<<512, 256, 0, stream>>>((float*)act, (size_t)2 * Bb * KPC / 2, 0.0f);
  fill_f32<<<256, 256, 0, stream>>>(cst, (size_t)Bb * NPC, 0.0f);
  fill_f32<<<256, 256, 0, stream>>>(hful, (size_t)Bb * NPC, 0.0f);
  copy_f32<<<4096, 256, 0, stream>>>(M0, Mw, (size_t)Bb * Nn * Wd);

  // ---- phase 1: encoder LSTM over T steps ----
  for (int t = 0; t < Tt; t++) {
    const bf16* hin  = henc + (size_t)(t & 1) * Bb * Wd;
    bf16*       hout = henc + (size_t)((t + 1) & 1) * Bb * Wd;
    enc_step_kernel<<<Wd / 16, 128, 0, stream>>>(
        xbf + (size_t)t * Bb * Wd, hin, Wenc, benc, cenc, hout,
        encs + (size_t)t * Bb * Wd);
  }

  // ---- phase 2: controller LSTM + external memory over T steps ----
  ctl_init_act<<<Bb, Wd, 0, stream>>>(encs, act);
  for (int t = 0; t < Tt; t++) {
    const bf16* ain  = act + (size_t)(t & 1) * Bb * KPC;
    bf16*       aout = act + (size_t)((t + 1) & 1) * Bb * KPC;
    ctl_step_kernel<<<NPC / 16, 128, 0, stream>>>(ain, Wc, bc, cst, hful, aout, out, t);
    mem_step_kernel<<<Bb, 128, 0, stream>>>(
        hful, Mw, encs + (size_t)(t + 1) * Bb * Wd, aout, (t + 1 < Tt) ? 1 : 0);
  }
}
